// LIIF_4767413698965
// MI455X (gfx1250) — matrix operand order repaired and checked by reading, not compiled, not run on hardware
//
#include <hip/hip_runtime.h>
#include <hip/hip_bf16.h>

// ---------------------------------------------------------------------------
// Types for CDNA5 WMMA (wave32): A/B = 16 bf16 per lane, C/D = 8 f32 per lane
// ---------------------------------------------------------------------------
typedef __attribute__((ext_vector_type(16))) __bf16 v16bf;
typedef __attribute__((ext_vector_type(8)))  float  v8f;
typedef __attribute__((ext_vector_type(4)))  unsigned int v4u;
typedef __attribute__((ext_vector_type(8)))  int v8i_t;
typedef __attribute__((ext_vector_type(4)))  int v4i_t;

#if __has_builtin(__builtin_amdgcn_tensor_load_to_lds) && \
    __has_builtin(__builtin_amdgcn_s_wait_tensorcnt)
#define USE_TDM 1
#endif

union FragAB {
  uint4 u[2];
  v16bf v;
};

__device__ __forceinline__ unsigned short f2bf(float f) {
  unsigned int u = __float_as_uint(f);
  u += 0x7FFFu + ((u >> 16) & 1u);      // round to nearest even
  return (unsigned short)(u >> 16);
}

// Async memory->LDS DMA (ASYNCcnt).  %0 = LDS byte offset (wave-relative,
// low 32 bits of a generic LDS pointer), %1 = 64-bit global address.
__device__ __forceinline__ void async_ld_b128(unsigned ldsOff, const void* g) {
  asm volatile("global_load_async_to_lds_b128 %0, %1, off"
               :: "v"(ldsOff), "v"(g) : "memory");
}
__device__ __forceinline__ void wait_async0() {
  asm volatile("s_wait_asynccnt 0x0" ::: "memory");
}

// ---------------------------------------------------------------------------
// Constants
// ---------------------------------------------------------------------------
#define BQ   16384      // B*Q = 2*8192
#define NPIX 8192       // B*H*W

// ===========================================================================
// Weight / input conversion kernels (bf16 re-layouts, all L2-resident)
// ===========================================================================
__global__ void cvt_inp(const float* __restrict__ inp, unsigned short* __restrict__ a0) {
  int idx = blockIdx.x * blockDim.x + threadIdx.x;      // NPIX*32
  int p = idx >> 5, ci = idx & 31;
  int b = p >> 12, hw = p & 4095;
  float v = (ci < 3) ? inp[((b * 3 + ci) << 12) + hw] : 0.0f;
  a0[idx] = f2bf(v);
}

__global__ void cvt_convw(const float* __restrict__ w, unsigned short* __restrict__ out,
                          int Cout, int Cin, int CinP) {
  int idx = blockIdx.x * blockDim.x + threadIdx.x;      // Cout*9*CinP
  int total = Cout * 9 * CinP;
  if (idx >= total) return;
  int n = idx / (9 * CinP);
  int rem = idx - n * (9 * CinP);
  int t = rem / CinP, ci = rem - t * CinP;
  int kh = t / 3, kw = t - kh * 3;
  float v = (ci < Cin) ? w[((n * Cin + ci) * 3 + kh) * 3 + kw] : 0.0f;
  out[idx] = f2bf(v);
}

__global__ void cvt_w0(const float* __restrict__ w0, unsigned short* __restrict__ w0t) {
  int idx = blockIdx.x * blockDim.x + threadIdx.x;      // 256*2304
  int n = idx / 2304, r = idx - n * 2304;
  int t = r >> 8, c = r & 255;
  w0t[idx] = f2bf(w0[(c * 9 + t) * 256 + n]);           // k' = t*256 + c
}

__global__ void cvt_wcell(const float* __restrict__ w0, float* __restrict__ wcell) {
  int idx = blockIdx.x * blockDim.x + threadIdx.x;      // 512
  int j = idx >> 8, n = idx & 255;
  wcell[idx] = w0[(2304 + j) * 256 + n];
}

__global__ void cvt_wsq(const float* __restrict__ w, unsigned short* __restrict__ wt) {
  int idx = blockIdx.x * blockDim.x + threadIdx.x;      // 256*256
  int n = idx >> 8, k = idx & 255;
  wt[idx] = f2bf(w[k * 256 + n]);                       // [N][K]
}

__global__ void cvt_w4(const float* __restrict__ w4, unsigned short* __restrict__ wt) {
  int idx = blockIdx.x * blockDim.x + threadIdx.x;      // 16*256
  int n = idx >> 8, k = idx & 255;
  wt[idx] = (n < 3) ? f2bf(w4[k * 3 + n]) : (unsigned short)0;
}

// Zero fill: featT OOB row (pixel index 8192) + generic 512B zero row
__global__ void zero_tail(unsigned short* __restrict__ a4zero,
                          unsigned short* __restrict__ zr) {
  int i = threadIdx.x;                                  // 256
  a4zero[i] = 0;
  zr[i] = 0;
}

// ===========================================================================
// Sampling prep: nearest-neighbor indices + local-ensemble weights per query
// ===========================================================================
__global__ void sample_prep(const float* __restrict__ coord, const float* __restrict__ cell,
                            int* __restrict__ cidx, float* __restrict__ cwgt,
                            float* __restrict__ rcout) {
  int bq = blockIdx.x * blockDim.x + threadIdx.x;
  if (bq >= BQ) return;
  float c0 = coord[bq * 2 + 0], c1 = coord[bq * 2 + 1];
  float ce0 = cell[bq * 2 + 0], ce1 = cell[bq * 2 + 1];
  rcout[bq * 2 + 0] = ce0 * 64.0f;
  rcout[bq * 2 + 1] = ce1 * 64.0f;
  const float rx = 1.0f / 64.0f;
  float area[4];
  int packed[4];
  int j = 0;
  #pragma unroll
  for (int vx = -1; vx <= 1; vx += 2) {
    #pragma unroll
    for (int vy = -1; vy <= 1; vy += 2) {
      float cy = fminf(fmaxf(c0 + vx * rx + 1e-6f, -1.0f + 1e-6f), 1.0f - 1e-6f);
      float cx = fminf(fmaxf(c1 + vy * rx + 1e-6f, -1.0f + 1e-6f), 1.0f - 1e-6f);
      int iy = (int)rintf(((cy + 1.0f) * 64.0f - 1.0f) * 0.5f);
      int ix = (int)rintf(((cx + 1.0f) * 64.0f - 1.0f) * 0.5f);
      iy = min(63, max(0, iy));
      ix = min(63, max(0, ix));
      float ly = -1.0f + (2.0f / 63.0f) * (float)iy;    // linspace(-1,1,64)
      float lx = -1.0f + (2.0f / 63.0f) * (float)ix;
      float r0 = (c0 - ly) * 64.0f;
      float r1 = (c1 - lx) * 64.0f;
      area[j] = fabsf(r0 * r1) + 1e-9f;
      packed[j] = iy * 64 + ix;
      ++j;
    }
  }
  float tot = area[0] + area[1] + area[2] + area[3];
  #pragma unroll
  for (int k = 0; k < 4; ++k) {
    cidx[k * BQ + bq] = packed[k];
    cwgt[k * BQ + bq] = area[3 - k] / tot;              // local-ensemble diagonal swap
  }
}

// ===========================================================================
// Implicit-GEMM conv (3x3 SAME + bias + ReLU), channels-last bf16 activations.
// A tile staged by GLOBAL_LOAD_ASYNC_TO_LDS_B128 (ASYNCcnt) per tap.
// ===========================================================================
template <int CINP, int COUT>
__global__ __launch_bounds__(256) void conv_gemm(
    const unsigned short* __restrict__ actIn,   // [NPIX][CINP] bf16
    const unsigned short* __restrict__ wT,      // [COUT][9*CINP] bf16
    const float* __restrict__ bias,             // [COUT] f32
    unsigned short* __restrict__ actOut,        // [NPIX][COUT] bf16
    const unsigned short* __restrict__ zrow)    // >=CINP bf16 zeros
{
  constexpr int NB = COUT / 16;
  constexpr int NBW = NB / 2;                   // n-blocks per wave
  __shared__ __align__(16) unsigned short sA[64 * CINP];

  const int tid = threadIdx.x;
  const int wid = tid >> 5, lane = tid & 31;
  const int mb = wid & 3, nh = wid >> 2;
  const int l15 = lane & 15, l16 = lane >> 4;
  const int tile = blockIdx.x;                  // 128 tiles of 64 pixels

  v8f acc[NBW];
  #pragma unroll
  for (int j = 0; j < NBW; ++j) acc[j] = v8f{0, 0, 0, 0, 0, 0, 0, 0};

  for (int t = 0; t < 9; ++t) {
    const int dy = t / 3 - 1, dx = t % 3 - 1;
    // ---- async-stage 64 x CINP tap tile into LDS (zero row at borders) ----
    {
      const int m = tid >> 2, tr = tid & 3;     // 4 threads per row
      const int p = tile * 64 + m;
      const int b = p >> 12, yy = (p >> 6) & 63, xx = p & 63;
      const int ny = yy + dy, nx = xx + dx;
      const bool ok = ((unsigned)ny < 64u) && ((unsigned)nx < 64u);
      const unsigned short* srcp = ok
          ? actIn + (((b << 12) + ny * 64 + nx) * CINP) + tr * (CINP / 4)
          : zrow + tr * (CINP / 4);
      const unsigned ldsOff = (unsigned)(size_t)&sA[m * CINP + tr * (CINP / 4)];
      #pragma unroll
      for (int v = 0; v < CINP / 32; ++v)
        async_ld_b128(ldsOff + v * 16, (const char*)srcp + v * 16);
      if (t < 8)
        __builtin_prefetch(wT + ((nh * NBW) << 4) * (9 * CINP) + (t + 1) * CINP, 0, 1);
    }
    wait_async0();
    __syncthreads();
    // ---- WMMA over this tap's K chunks ----
    #pragma unroll
    for (int c = 0; c < CINP / 32; ++c) {
      FragAB a;
      const unsigned short* ar = &sA[(mb * 16 + l15) * CINP + c * 32 + l16 * 8];
      a.u[0] = *(const uint4*)ar;
      a.u[1] = *(const uint4*)(ar + 16);
      #pragma unroll
      for (int j = 0; j < NBW; ++j) {
        const int n = ((nh * NBW + j) << 4) + l15;
        const unsigned short* br = wT + n * (9 * CINP) + t * CINP + c * 32 + l16 * 8;
        FragAB bb;
        bb.u[0] = *(const uint4*)br;
        bb.u[1] = *(const uint4*)(br + 16);
        acc[j] = __builtin_amdgcn_wmma_f32_16x16x32_bf16(
            false, a.v, false, bb.v, (short)0, acc[j], false, false);
      }
    }
    __syncthreads();
  }

  // ---- epilogue: bias + ReLU -> bf16 channels-last ----
  #pragma unroll
  for (int j = 0; j < NBW; ++j) {
    const int n = ((nh * NBW + j) << 4) + l15;
    const float bs = bias[n];
    #pragma unroll
    for (int r = 0; r < 8; ++r) {
      const int m = mb * 16 + l16 * 8 + r;
      float v = acc[j][r] + bs;
      v = v > 0.0f ? v : 0.0f;
      actOut[(tile * 64 + m) * COUT + n] = f2bf(v);
    }
  }
}

// ===========================================================================
// Fused LIIF MLP: gather(9 taps x 256ch) -> [2306->256->256->256->256->3]
// Tap tiles gathered by the Tensor Data Mover (gather mode, 16-bit row
// indices into featT[8193][256]; row 8192 is the zero row). Rows r = bq*4+cr.
// ===========================================================================
__global__ __launch_bounds__(256) void mlp_fused(
    const unsigned short* __restrict__ featT,   // [NPIX+1][256] bf16 (conv4 out + zero row)
    const unsigned short* __restrict__ w0t,     // [256][2304] bf16, k'=t*256+c
    const float* __restrict__ wcell,            // [2][256] f32 (rows 2304/2305 of w0)
    const unsigned short* __restrict__ w1t,     // [256][256]
    const unsigned short* __restrict__ w2t,
    const unsigned short* __restrict__ w3t,
    const unsigned short* __restrict__ w4t,     // [16][256] (N padded 3->16)
    const float* __restrict__ b0, const float* __restrict__ b1,
    const float* __restrict__ b2, const float* __restrict__ b3,
    const float* __restrict__ b4,
    const int* __restrict__ cidx,               // [4][BQ] packed iy*64+ix
    const float* __restrict__ cwgt,             // [4][BQ] ensemble weight
    const float* __restrict__ rc,               // [BQ][2] rel_cell
    float* __restrict__ out)                    // [BQ][3] f32
{
  __shared__ __align__(16) unsigned short sB[64 * 256];
  __shared__ int   sPix[64];
  __shared__ float sW[64], sC0[64], sC1[64];

  const int tid = threadIdx.x;
  const int wid = tid >> 5, lane = tid & 31;
  const int mb = wid & 3, nhalf = wid >> 2;
  const int l15 = lane & 15, l16 = lane >> 4;
  const int tile = blockIdx.x;                  // 1024 tiles of 64 rows

  if (tid < 64) {
    const int r = tile * 64 + tid;
    const int bq = r >> 2, cr = r & 3;
    sPix[tid] = cidx[cr * BQ + bq];
    sW[tid]   = cwgt[cr * BQ + bq];
    sC0[tid]  = rc[bq * 2 + 0];
    sC1[tid]  = rc[bq * 2 + 1];
  }
  __syncthreads();

  const unsigned ldsBase = (unsigned)(size_t)&sB[0];

  v8f acc[8];
  #pragma unroll
  for (int j = 0; j < 8; ++j) acc[j] = v8f{0, 0, 0, 0, 0, 0, 0, 0};

  // ---------------- layer 0: K = 9 taps x 256 ----------------
  for (int t = 0; t < 9; ++t) {
    const int dy = t / 3 - 1, dx = t % 3 - 1;
    if (t < 8)
      __builtin_prefetch(w0t + (((nhalf * 8) << 4) + l15) * 2304 + (t + 1) * 256, 0, 1);
#if defined(USE_TDM)
    // ---- TDM gather: waves 0..3 each fetch 16 rows (512B each) into LDS ----
    if (wid < 4) {
      unsigned dpack[8];
      #pragma unroll
      for (int j2 = 0; j2 < 8; ++j2) {
        unsigned lohi[2];
        #pragma unroll
        for (int h = 0; h < 2; ++h) {
          const int m = wid * 16 + j2 * 2 + h;
          const int pk = sPix[m];
          const int iy = pk >> 6, ix = pk & 63;
          const int ny = iy + dy, nx = ix + dx;
          const int bq = (tile * 64 + m) >> 2;
          const int b = bq >> 13;               // Q = 8192
          const bool ok = ((unsigned)ny < 64u) && ((unsigned)nx < 64u);
          lohi[h] = ok ? (unsigned)((b << 12) + ny * 64 + nx) : 8192u;  // zero row
        }
        dpack[j2] = lohi[0] | (lohi[1] << 16);
      }
      const unsigned long long ga = (unsigned long long)(size_t)featT;
      v4u g0;
      g0[0] = 0x80000001u;                      // count=1, gather_mode=1, 16-bit idx
      g0[1] = (unsigned)__builtin_amdgcn_readfirstlane(
                  (int)(ldsBase + (unsigned)(wid * 16 * 512)));
      g0[2] = (unsigned)(ga & 0xFFFFFFFFu);
      g0[3] = (unsigned)((ga >> 32) & 0x01FFFFFFu) | 0x80000000u;  // type=2
      v8i_t g1;
      g1[0] = 0x00010000;                       // data_size = 2 bytes
      g1[1] = 0x01000000;                       // tensor_dim0 = 256
      g1[2] = 0x20010000;                       // tensor_dim1 = 8193
      g1[3] = 0x01000000;                       // tile_dim0   = 256
      g1[4] = 16;                               // tile_dim1   = 16 indices
      g1[5] = 256;                              // tensor_dim0_stride = 256
      g1[6] = 0;
      g1[7] = 0;
      v4i_t g2, g3;
      #pragma unroll
      for (int j2 = 0; j2 < 4; ++j2) {
        g2[j2] = __builtin_amdgcn_readfirstlane((int)dpack[j2]);
        g3[j2] = __builtin_amdgcn_readfirstlane((int)dpack[4 + j2]);
      }
      v8i_t g4 = v8i_t{0, 0, 0, 0, 0, 0, 0, 0}; // VADDR4 unused (NULL per ISA)
      __builtin_amdgcn_tensor_load_to_lds(g0, g1, g2, g3, g4, 0);
    }
    __builtin_amdgcn_s_wait_tensorcnt(0);
#else
    // ---- fallback: per-lane async memory->LDS copies ----
    {
      const int m = tid >> 2, tr = tid & 3;
      const int pk = sPix[m];
      const int iy = pk >> 6, ix = pk & 63;
      const int ny = iy + dy, nx = ix + dx;
      const int bq = (tile * 64 + m) >> 2;
      const int b = bq >> 13;
      const bool ok = ((unsigned)ny < 64u) && ((unsigned)nx < 64u);
      const int pix = ok ? ((b << 12) + ny * 64 + nx) : 8192;   // zero row
      const unsigned short* srcp = featT + (pix << 8) + tr * 64;
      const unsigned ldsOff = ldsBase + (unsigned)(tid * 128);  // m*512 + tr*128
      #pragma unroll
      for (int v = 0; v < 8; ++v)
        async_ld_b128(ldsOff + v * 16, (const char*)srcp + v * 16);
    }
    wait_async0();
#endif
    __syncthreads();
    #pragma unroll
    for (int c = 0; c < 8; ++c) {
      FragAB a;
      const unsigned short* ar = &sB[((mb * 16 + l15) << 8) + c * 32 + l16 * 8];
      a.u[0] = *(const uint4*)ar;
      a.u[1] = *(const uint4*)(ar + 16);
      #pragma unroll
      for (int j = 0; j < 8; ++j) {
        const int n = ((nhalf * 8 + j) << 4) + l15;
        const unsigned short* br = w0t + n * 2304 + t * 256 + c * 32 + l16 * 8;
        FragAB bb;
        bb.u[0] = *(const uint4*)br;
        bb.u[1] = *(const uint4*)(br + 16);
        acc[j] = __builtin_amdgcn_wmma_f32_16x16x32_bf16(
            false, a.v, false, bb.v, (short)0, acc[j], false, false);
      }
    }
    __syncthreads();
  }

  // layer-0 epilogue: + bias + rel_cell features, ReLU -> h in LDS
  #pragma unroll
  for (int j = 0; j < 8; ++j) {
    const int n = ((nhalf * 8 + j) << 4) + l15;
    const float bs = b0[n], wc0 = wcell[n], wc1 = wcell[256 + n];
    #pragma unroll
    for (int r = 0; r < 8; ++r) {
      const int m = mb * 16 + l16 * 8 + r;
      float v = acc[j][r] + bs + sC0[m] * wc0 + sC1[m] * wc1;
      v = v > 0.0f ? v : 0.0f;
      sB[(m << 8) + n] = f2bf(v);
    }
  }
  __syncthreads();

  // ---------------- layers 1..3: 256x256, in-place in LDS ----------------
  const unsigned short* Ws[3] = {w1t, w2t, w3t};
  const float* Bs[3] = {b1, b2, b3};
  for (int l = 0; l < 3; ++l) {
    v8f a2[8];
    #pragma unroll
    for (int j = 0; j < 8; ++j) a2[j] = v8f{0, 0, 0, 0, 0, 0, 0, 0};
    #pragma unroll
    for (int c = 0; c < 8; ++c) {
      FragAB a;
      const unsigned short* ar = &sB[((mb * 16 + l15) << 8) + c * 32 + l16 * 8];
      a.u[0] = *(const uint4*)ar;
      a.u[1] = *(const uint4*)(ar + 16);
      #pragma unroll
      for (int j = 0; j < 8; ++j) {
        const int n = ((nhalf * 8 + j) << 4) + l15;
        const unsigned short* br = Ws[l] + n * 256 + c * 32 + l16 * 8;
        FragAB bb;
        bb.u[0] = *(const uint4*)br;
        bb.u[1] = *(const uint4*)(br + 16);
        a2[j] = __builtin_amdgcn_wmma_f32_16x16x32_bf16(
            false, a.v, false, bb.v, (short)0, a2[j], false, false);
      }
    }
    __syncthreads();                            // all reads of sB done
    #pragma unroll
    for (int j = 0; j < 8; ++j) {
      const int n = ((nhalf * 8 + j) << 4) + l15;
      const float bs = Bs[l][n];
      #pragma unroll
      for (int r = 0; r < 8; ++r) {
        const int m = mb * 16 + l16 * 8 + r;
        float v = a2[j][r] + bs;
        v = v > 0.0f ? v : 0.0f;
        sB[(m << 8) + n] = f2bf(v);
      }
    }
    __syncthreads();
  }

  // ---------------- layer 4: 256 -> 3 (padded to 16) + ensemble ----------
  if (wid < 4) {                                // wave-uniform, EXEC all ones
    const int mb4 = wid;
    v8f a4 = v8f{0, 0, 0, 0, 0, 0, 0, 0};
    #pragma unroll
    for (int c = 0; c < 8; ++c) {
      FragAB a;
      const unsigned short* ar = &sB[((mb4 * 16 + l15) << 8) + c * 32 + l16 * 8];
      a.u[0] = *(const uint4*)ar;
      a.u[1] = *(const uint4*)(ar + 16);
      const unsigned short* br = w4t + l15 * 256 + c * 32 + l16 * 8;
      FragAB bb;
      bb.u[0] = *(const uint4*)br;
      bb.u[1] = *(const uint4*)(br + 16);
      a4 = __builtin_amdgcn_wmma_f32_16x16x32_bf16(
          false, a.v, false, bb.v, (short)0, a4, false, false);
    }
    const int n = l15;
    if (n < 3) {
      const float bs = b4[n];
      const int mbase = mb4 * 16 + l16 * 8;     // 8 consecutive rows = 2 queries
      float s0 = 0.0f, s1 = 0.0f;
      #pragma unroll
      for (int r = 0; r < 4; ++r) s0 += (a4[r] + bs) * sW[mbase + r];
      #pragma unroll
      for (int r = 4; r < 8; ++r) s1 += (a4[r] + bs) * sW[mbase + r];
      const int bq0 = (tile * 64 + mbase) >> 2;
      out[bq0 * 3 + n] = s0;
      out[(bq0 + 1) * 3 + n] = s1;
    }
  }
}

// ===========================================================================
// Host launcher
// ===========================================================================
extern "C" void kernel_launch(void* const* d_in, const int* in_sizes, int n_in,
                              void* d_out, int out_size, void* d_ws, size_t ws_size,
                              hipStream_t stream) {
  const float* inp   = (const float*)d_in[0];
  const float* coord = (const float*)d_in[1];
  const float* cell  = (const float*)d_in[2];
  const float* c1w = (const float*)d_in[3];  const float* c1b = (const float*)d_in[4];
  const float* c2w = (const float*)d_in[5];  const float* c2b = (const float*)d_in[6];
  const float* c3w = (const float*)d_in[7];  const float* c3b = (const float*)d_in[8];
  const float* c4w = (const float*)d_in[9];  const float* c4b = (const float*)d_in[10];
  const float* w0  = (const float*)d_in[11]; const float* b0  = (const float*)d_in[12];
  const float* w1  = (const float*)d_in[13]; const float* b1  = (const float*)d_in[14];
  const float* w2  = (const float*)d_in[15]; const float* b2  = (const float*)d_in[16];
  const float* w3  = (const float*)d_in[17]; const float* b3  = (const float*)d_in[18];
  const float* w4  = (const float*)d_in[19]; const float* b4  = (const float*)d_in[20];
  float* out = (float*)d_out;

  char* ws = (char*)d_ws;
  size_t o = 0;
  auto alloc = [&](size_t bytes) -> char* {
    char* p = ws + o;
    o = (o + bytes + 255) & ~(size_t)255;
    return p;
  };
  unsigned short* A0  = (unsigned short*)alloc((size_t)NPIX * 32 * 2);
  unsigned short* A1  = (unsigned short*)alloc((size_t)NPIX * 64 * 2);
  unsigned short* A2  = (unsigned short*)alloc((size_t)NPIX * 128 * 2);
  unsigned short* A3  = (unsigned short*)alloc((size_t)NPIX * 256 * 2);
  unsigned short* A4  = (unsigned short*)alloc((size_t)(NPIX + 1) * 256 * 2); // + zero row
  unsigned short* C1W = (unsigned short*)alloc((size_t)64 * 288 * 2);
  unsigned short* C2W = (unsigned short*)alloc((size_t)128 * 576 * 2);
  unsigned short* C3W = (unsigned short*)alloc((size_t)256 * 1152 * 2);
  unsigned short* C4W = (unsigned short*)alloc((size_t)256 * 2304 * 2);
  unsigned short* W0T = (unsigned short*)alloc((size_t)256 * 2304 * 2);
  float*          WCL = (float*)alloc((size_t)512 * 4);
  unsigned short* W1T = (unsigned short*)alloc((size_t)256 * 256 * 2);
  unsigned short* W2T = (unsigned short*)alloc((size_t)256 * 256 * 2);
  unsigned short* W3T = (unsigned short*)alloc((size_t)256 * 256 * 2);
  unsigned short* W4T = (unsigned short*)alloc((size_t)16 * 256 * 2);
  unsigned short* ZR  = (unsigned short*)alloc((size_t)256 * 2);             // 512B zeros
  int*            CIX = (int*)alloc((size_t)4 * BQ * 4);
  float*          CWG = (float*)alloc((size_t)4 * BQ * 4);
  float*          RC  = (float*)alloc((size_t)BQ * 2 * 4);

  // ---- prep ----
  zero_tail<<<1, 256, 0, stream>>>(A4 + (size_t)NPIX * 256, ZR);
  cvt_inp<<<(NPIX * 32) / 256, 256, 0, stream>>>(inp, A0);
  cvt_convw<<<(64 * 9 * 32) / 256, 256, 0, stream>>>(c1w, C1W, 64, 3, 32);
  cvt_convw<<<(128 * 9 * 64) / 256, 256, 0, stream>>>(c2w, C2W, 128, 64, 64);
  cvt_convw<<<(256 * 9 * 128) / 256, 256, 0, stream>>>(c3w, C3W, 256, 128, 128);
  cvt_convw<<<(256 * 9 * 256) / 256, 256, 0, stream>>>(c4w, C4W, 256, 256, 256);
  cvt_w0<<<(256 * 2304) / 256, 256, 0, stream>>>(w0, W0T);
  cvt_wcell<<<2, 256, 0, stream>>>(w0, WCL);
  cvt_wsq<<<256, 256, 0, stream>>>(w1, W1T);
  cvt_wsq<<<256, 256, 0, stream>>>(w2, W2T);
  cvt_wsq<<<256, 256, 0, stream>>>(w3, W3T);
  cvt_w4<<<16, 256, 0, stream>>>(w4, W4T);
  sample_prep<<<BQ / 256, 256, 0, stream>>>(coord, cell, CIX, CWG, RC);

  // ---- encoder (implicit-GEMM WMMA convs, async LDS staging) ----
  conv_gemm<32, 64><<<128, 256, 0, stream>>>(A0, C1W, c1b, A1, ZR);
  conv_gemm<64, 128><<<128, 256, 0, stream>>>(A1, C2W, c2b, A2, ZR);
  conv_gemm<128, 256><<<128, 256, 0, stream>>>(A2, C3W, c3b, A3, ZR);
  conv_gemm<256, 256><<<128, 256, 0, stream>>>(A3, C4W, c4b, A4, ZR);

  // ---- fused LIIF MLP (TDM gather) + local ensemble ----
  mlp_fused<<<(4 * BQ) / 64, 256, 0, stream>>>(
      A4, W0T, WCL, W1T, W2T, W3T, W4T,
      b0, b1, b2, b3, b4, CIX, CWG, RC, out);
}
